// ADMETPredictor_78005196030508
// MI455X (gfx1250) — matrix operand order repaired and checked by reading, not compiled, not run on hardware
//
#include <hip/hip_runtime.h>
#include <hip/hip_bf16.h>
#include <math.h>

// ---------------------------------------------------------------------------
// ADMET GNN forward for MI455X (gfx1250, wave32, WMMA).
//   deg -> invdeg; cvt weights->bf16; h = relu(WMMA(x, W_in^T)+b)
//   3x { agg atomics; hb=bf16(h+agg*invdeg); tmp=WMMA(hb,W^T)+b (+BN stats);
//        bn finalize; h = relu(bn(tmp)) }
//   colsum -> head (pool + 9 sigmoids)
// GEMM epilogues transpose C through LDS for coalesced b128 stores.
// ---------------------------------------------------------------------------

#define H      256
#define FIN    148
#define KPAD   160     // FIN padded to multiple of 32 for K=32 bf16 WMMA
#define LAYERS 3
#define NTASKS 9
#define BN_EPS 1e-5f
#define LSTR   20      // LDS row stride (floats): banks for rows r / r+8 disjoint

typedef __attribute__((ext_vector_type(8)))  float   v8f;
typedef __attribute__((ext_vector_type(8)))  __bf16  v8bf;
typedef __attribute__((ext_vector_type(16))) __bf16  v16bf;
typedef __attribute__((ext_vector_type(4)))  __bf16  v4bf;

static __device__ __forceinline__ v16bf bf_cat(v8bf lo, v8bf hi) {
  v16bf r;
#pragma unroll
  for (int i = 0; i < 8; ++i) { r[i] = lo[i]; r[i + 8] = hi[i]; }
  return r;
}

// ---------------------------- utilities ------------------------------------
__global__ void k_zero4(float4* __restrict__ p, int n4) {
  int t = blockIdx.x * blockDim.x + threadIdx.x;
  if (t < n4) p[t] = make_float4(0.f, 0.f, 0.f, 0.f);
}

__global__ void k_deg(const int* __restrict__ row, float* __restrict__ deg, int E) {
  int t = blockIdx.x * blockDim.x + threadIdx.x;
  if (t < E) atomicAdd(&deg[row[t]], 1.0f);
}

__global__ void k_invdeg(float* __restrict__ deg, int N) {
  int t = blockIdx.x * blockDim.x + threadIdx.x;
  if (t < N) { float d = deg[t]; deg[t] = 1.0f / fmaxf(d, 1.0f); }
}

// f32 [rows, kin] row-major  ->  bf16 [rows, kout] row-major, zero padded.
__global__ void k_cvt_w(const float* __restrict__ src, __bf16* __restrict__ dst,
                        int rows, int kin, int kout) {
  int t = blockIdx.x * blockDim.x + threadIdx.x;
  if (t >= rows * kout) return;
  int o = t / kout, k = t - o * kout;
  dst[t] = (k < kin) ? (__bf16)src[o * kin + k] : (__bf16)0.0f;
}

// --------------------------- input projection ------------------------------
// h[N,256] = relu(x[N,148] @ W_in^T + b_in); A f32->bf16 on the fly.
// One wave owns a 16-node x 16-channel tile; o-tile = blockIdx.y*8+wave.
__global__ void k_gemm_in(const float* __restrict__ x, const __bf16* __restrict__ wb,
                          const float* __restrict__ bias, float* __restrict__ h,
                          int ntiles) {
  __shared__ float lds[8 * 16 * LSTR];
  const int lane = threadIdx.x & 31;
  const int wave = threadIdx.x >> 5;
  const int ln = lane & 15;
  const int hi = lane >> 4;
  const int o0 = (blockIdx.y * 8 + wave) * 16;
  float* wl = &lds[wave * 16 * LSTR];

  v16bf B[KPAD / 32];  // weights register-resident, reused across m strip
  {
    const __bf16* wr = wb + (size_t)(o0 + ln) * KPAD + hi * 8;
#pragma unroll
    for (int t = 0; t < KPAD / 32; ++t) {
      v8bf b0 = *(const v8bf*)(wr + t * 32);
      v8bf b1 = *(const v8bf*)(wr + t * 32 + 16);
      B[t] = bf_cat(b0, b1);
    }
  }
  const float bval = bias[o0 + ln];
  const int rr = lane >> 1;          // output row within tile for store phase
  const int seg = (lane & 1) * 8;    // 8-channel segment

  for (int mt = blockIdx.x; mt < ntiles; mt += gridDim.x) {
    const int m0 = mt * 16;
    const float* xr = x + (size_t)(m0 + ln) * FIN;
    v8f C = {};
#pragma unroll
    for (int t = 0; t < KPAD / 32; ++t) {
      v16bf A;
#pragma unroll
      for (int i = 0; i < 8; ++i) {
        int k0 = t * 32 + hi * 8 + i;
        int k1 = t * 32 + 16 + hi * 8 + i;
        A[i]     = (__bf16)((k0 < FIN) ? xr[k0] : 0.0f);
        A[i + 8] = (__bf16)((k1 < FIN) ? xr[k1] : 0.0f);
      }
      C = __builtin_amdgcn_wmma_f32_16x16x32_bf16(false, A, false, B[t],
                                                  (short)0, C, false, false);
    }
    // relu epilogue -> LDS transpose -> coalesced b128 stores
#pragma unroll
    for (int r = 0; r < 8; ++r)
      wl[(hi * 8 + r) * LSTR + ln] = fmaxf(C[r] + bval, 0.0f);
    __syncthreads();
    {
      const float* src = wl + rr * LSTR + seg;
      float4 a = *(const float4*)(src);
      float4 b = *(const float4*)(src + 4);
      float* dst = h + (size_t)(m0 + rr) * H + o0 + seg;
      *(float4*)(dst) = a;
      *(float4*)(dst + 4) = b;
    }
    __syncthreads();
  }
}

// --------------------------- neighbor aggregation --------------------------
// agg[row[e]] += h[col[e]]  (float4 loads, 4x global_atomic_add_f32)
__global__ void k_aggregate(const int* __restrict__ row, const int* __restrict__ col,
                            const float* __restrict__ h, float* __restrict__ agg,
                            int E) {
  long t = (long)blockIdx.x * blockDim.x + threadIdx.x;
  if (t >= (long)E * 64) return;
  int e = (int)(t >> 6);
  int c = ((int)t & 63) << 2;
  int s = col[e], d = row[e];
  float4 v = *(const float4*)(h + (size_t)s * H + c);
  float* p = agg + (size_t)d * H + c;
  atomicAdd(p + 0, v.x);
  atomicAdd(p + 1, v.y);
  atomicAdd(p + 2, v.z);
  atomicAdd(p + 3, v.w);
}

// hb = bf16(h + agg * invdeg)
__global__ void k_make_hb(const float* __restrict__ h, const float* __restrict__ agg,
                          const float* __restrict__ invdeg, __bf16* __restrict__ hb,
                          int N) {
  long t = (long)blockIdx.x * blockDim.x + threadIdx.x;
  if (t >= (long)N * 64) return;
  int n = (int)(t >> 6);
  int c = ((int)t & 63) << 2;
  float inv = invdeg[n];
  float4 hv = *(const float4*)(h + (size_t)n * H + c);
  float4 av = *(const float4*)(agg + (size_t)n * H + c);
  v4bf o;
  o[0] = (__bf16)(hv.x + av.x * inv);
  o[1] = (__bf16)(hv.y + av.y * inv);
  o[2] = (__bf16)(hv.z + av.z * inv);
  o[3] = (__bf16)(hv.w + av.w * inv);
  *(v4bf*)(hb + (size_t)n * H + c) = o;
}

// --------------------------- layer GEMM + BN stats -------------------------
// out[N,256] = hb[N,256] @ W^T + b ; per-channel sum/sumsq kept in registers
// across the wave's m strip, single atomicAdd pair per lane at end.
__global__ void k_gemm_layer(const __bf16* __restrict__ hb, const __bf16* __restrict__ wb,
                             const float* __restrict__ bias, float* __restrict__ out,
                             float* __restrict__ bnsum, float* __restrict__ bnsq,
                             int ntiles) {
  __shared__ float lds[8 * 16 * LSTR];
  const int lane = threadIdx.x & 31;
  const int wave = threadIdx.x >> 5;
  const int ln = lane & 15;
  const int hi = lane >> 4;
  const int o0 = (blockIdx.y * 8 + wave) * 16;
  float* wl = &lds[wave * 16 * LSTR];

  v16bf B[8];
  {
    const __bf16* wr = wb + (size_t)(o0 + ln) * H + hi * 8;
#pragma unroll
    for (int t = 0; t < 8; ++t) {
      v8bf b0 = *(const v8bf*)(wr + t * 32);
      v8bf b1 = *(const v8bf*)(wr + t * 32 + 16);
      B[t] = bf_cat(b0, b1);
    }
  }
  const float bval = bias[o0 + ln];
  const int rr = lane >> 1;
  const int seg = (lane & 1) * 8;
  float ssum = 0.f, ssq = 0.f;

  for (int mt = blockIdx.x; mt < ntiles; mt += gridDim.x) {
    const int m0 = mt * 16;
    const __bf16* ar = hb + (size_t)(m0 + ln) * H;
    int mn = mt + gridDim.x;
    if (mn < ntiles) {  // global_prefetch_b8 of next A strip row
      const __bf16* pr = hb + (size_t)(mn * 16 + ln) * H;
      __builtin_prefetch(pr, 0, 0);
      __builtin_prefetch(pr + 128, 0, 0);
    }
    v8f C = {};
#pragma unroll
    for (int t = 0; t < 8; ++t) {
      v8bf a0 = *(const v8bf*)(ar + t * 32 + hi * 8);
      v8bf a1 = *(const v8bf*)(ar + t * 32 + 16 + hi * 8);
      C = __builtin_amdgcn_wmma_f32_16x16x32_bf16(false, bf_cat(a0, a1), false, B[t],
                                                  (short)0, C, false, false);
    }
    // bias + BN stats, then LDS transpose -> coalesced b128 stores
#pragma unroll
    for (int r = 0; r < 8; ++r) {
      float v = C[r] + bval;
      ssum += v;
      ssq += v * v;
      wl[(hi * 8 + r) * LSTR + ln] = v;
    }
    __syncthreads();
    {
      const float* src = wl + rr * LSTR + seg;
      float4 a = *(const float4*)(src);
      float4 b = *(const float4*)(src + 4);
      float* dst = out + (size_t)(m0 + rr) * H + o0 + seg;
      *(float4*)(dst) = a;
      *(float4*)(dst + 4) = b;
    }
    __syncthreads();
  }
  atomicAdd(&bnsum[o0 + ln], ssum);
  atomicAdd(&bnsq[o0 + ln], ssq);
}

// --------------------------- batchnorm -------------------------------------
__global__ void k_bn_finalize(const float* __restrict__ bnsum, const float* __restrict__ bnsq,
                              const float* __restrict__ gamma, const float* __restrict__ beta,
                              float* __restrict__ scale, float* __restrict__ shift,
                              float invN) {
  int c = threadIdx.x;
  float mu = bnsum[c] * invN;
  float var = fmaxf(bnsq[c] * invN - mu * mu, 0.0f);
  float sc = gamma[c] * rsqrtf(var + BN_EPS);
  scale[c] = sc;
  shift[c] = beta[c] - mu * sc;
}

__global__ void k_bn_apply(const float* __restrict__ tmp, const float* __restrict__ scale,
                           const float* __restrict__ shift, float* __restrict__ h, int N) {
  long t = (long)blockIdx.x * blockDim.x + threadIdx.x;
  if (t >= (long)N * 64) return;
  int n = (int)(t >> 6);
  int c = ((int)t & 63) << 2;
  float4 v = *(const float4*)(tmp + (size_t)n * H + c);
  float4 o;
  o.x = fmaxf(v.x * scale[c + 0] + shift[c + 0], 0.f);
  o.y = fmaxf(v.y * scale[c + 1] + shift[c + 1], 0.f);
  o.z = fmaxf(v.z * scale[c + 2] + shift[c + 2], 0.f);
  o.w = fmaxf(v.w * scale[c + 3] + shift[c + 3], 0.f);
  *(float4*)(h + (size_t)n * H + c) = o;
}

// --------------------------- readout + heads -------------------------------
__global__ void k_colsum(const float* __restrict__ h, float* __restrict__ gsum, int N) {
  int c = threadIdx.x;
  float s = 0.f;
  for (int n = blockIdx.x; n < N; n += gridDim.x) s += h[(size_t)n * H + c];
  atomicAdd(&gsum[c], s);
}

__global__ void k_head(const float* __restrict__ gsum, const float* __restrict__ Wp,
                       const float* __restrict__ bp, const float* __restrict__ Wh,
                       const float* __restrict__ bh, float* __restrict__ out, float invN) {
  __shared__ float g[H];
  __shared__ float p[H];
  int c = threadIdx.x;
  g[c] = gsum[c] * invN;
  __syncthreads();
  float s = bp[c];
#pragma unroll 8
  for (int k = 0; k < H; ++k) s += g[k] * Wp[c * H + k];
  p[c] = fmaxf(s, 0.f);
  __syncthreads();
  if (c < NTASKS) {
    float s2 = bh[c];
#pragma unroll 8
    for (int k = 0; k < H; ++k) s2 += p[k] * Wh[c * H + k];
    out[c] = 1.0f / (1.0f + __expf(-s2));
  }
}

// ---------------------------------------------------------------------------
extern "C" void kernel_launch(void* const* d_in, const int* in_sizes, int n_in,
                              void* d_out, int out_size, void* d_ws, size_t ws_size,
                              hipStream_t stream) {
  (void)n_in; (void)out_size; (void)ws_size;
  const float* x       = (const float*)d_in[0];
  const int*   ei      = (const int*)d_in[1];
  const float* W_in    = (const float*)d_in[2];
  const float* b_in    = (const float*)d_in[3];
  const float* W_gnn   = (const float*)d_in[4];
  const float* b_gnn   = (const float*)d_in[5];
  const float* gamma   = (const float*)d_in[6];
  const float* beta    = (const float*)d_in[7];
  const float* W_pool  = (const float*)d_in[8];
  const float* b_pool  = (const float*)d_in[9];
  const float* W_heads = (const float*)d_in[10];
  const float* b_heads = (const float*)d_in[11];
  float* out = (float*)d_out;

  const int N = in_sizes[0] / FIN;   // 200000
  const int E = in_sizes[1] / 2;     // 800000
  const int ntiles = N / 16;         // 12500
  const float invN = 1.0f / (float)N;

  // workspace carve-out (256B aligned)
  char* base = (char*)d_ws;
  size_t off = 0;
  auto take = [&](size_t bytes) -> void* {
    void* p = base + off;
    off = (off + bytes + 255) & ~(size_t)255;
    return p;
  };
  float*  h     = (float*)take((size_t)N * H * 4);     // activations
  float*  tmp   = (float*)take((size_t)N * H * 4);     // agg, then pre-BN GEMM out
  __bf16* hb    = (__bf16*)take((size_t)N * H * 2);    // bf16 GEMM input
  float*  deg   = (float*)take((size_t)N * 4);         // deg -> invdeg (in place)
  __bf16* wbin  = (__bf16*)take((size_t)H * KPAD * 2);
  __bf16* wbg   = (__bf16*)take((size_t)LAYERS * H * H * 2);
  float*  bnsum = (float*)take(H * 4);
  float*  bnsq  = (float*)take(H * 4);
  float*  bnsc  = (float*)take(H * 4);
  float*  bnsh  = (float*)take(H * 4);
  float*  gsum  = (float*)take(H * 4);

  const int row_off = 0, col_off = E;
  const int BLK = 256;
  const int GX = 1024;              // m-strips per gemm; grid.y=2 covers 16 o-tiles
  const int nh4 = N * H / 4;

  // degrees
  k_zero4<<<(N / 4 + BLK - 1) / BLK, BLK, 0, stream>>>((float4*)deg, N / 4);
  k_deg<<<(E + BLK - 1) / BLK, BLK, 0, stream>>>(ei + row_off, deg, E);
  k_invdeg<<<(N + BLK - 1) / BLK, BLK, 0, stream>>>(deg, N);

  // weight conversion to bf16 (resident in 192MB L2 afterwards)
  k_cvt_w<<<(H * KPAD + BLK - 1) / BLK, BLK, 0, stream>>>(W_in, wbin, H, FIN, KPAD);
  for (int l = 0; l < LAYERS; ++l)
    k_cvt_w<<<(H * H + BLK - 1) / BLK, BLK, 0, stream>>>(W_gnn + (size_t)l * H * H,
                                                         wbg + (size_t)l * H * H,
                                                         H, H, H);

  // input projection
  k_gemm_in<<<dim3(GX, 2), BLK, 0, stream>>>(x, wbin, b_in, h, ntiles);

  // GNN layers
  for (int l = 0; l < LAYERS; ++l) {
    k_zero4<<<(nh4 + BLK - 1) / BLK, BLK, 0, stream>>>((float4*)tmp, nh4);  // agg=0
    k_aggregate<<<(int)(((long)E * 64 + BLK - 1) / BLK), BLK, 0, stream>>>(
        ei + row_off, ei + col_off, h, tmp, E);
    k_make_hb<<<(int)(((long)N * 64 + BLK - 1) / BLK), BLK, 0, stream>>>(h, tmp, deg, hb, N);
    k_zero4<<<1, BLK, 0, stream>>>((float4*)bnsum, H / 4);
    k_zero4<<<1, BLK, 0, stream>>>((float4*)bnsq, H / 4);
    k_gemm_layer<<<dim3(GX, 2), BLK, 0, stream>>>(hb, wbg + (size_t)l * H * H,
                                                  b_gnn + (size_t)l * H, tmp,
                                                  bnsum, bnsq, ntiles);
    k_bn_finalize<<<1, H, 0, stream>>>(bnsum, bnsq, gamma + (size_t)l * H,
                                       beta + (size_t)l * H, bnsc, bnsh, invN);
    k_bn_apply<<<(int)(((long)N * 64 + BLK - 1) / BLK), BLK, 0, stream>>>(tmp, bnsc, bnsh, h, N);
  }

  // readout + heads
  k_zero4<<<1, BLK, 0, stream>>>((float4*)gsum, H / 4);
  k_colsum<<<512, H, 0, stream>>>(h, gsum, N);
  k_head<<<1, H, 0, stream>>>(gsum, W_pool, b_pool, W_heads, b_heads, out, invN);
}
